// MultiHeadAttention_18580028523065
// MI455X (gfx1250) — matrix operand order repaired and checked by reading, not compiled, not run on hardware
//
#include <hip/hip_runtime.h>
#include <hip/hip_bf16.h>

#define EMBED 768
#define NHEAD 12
#define HDIM  64
#define BATCH 16
#define SEQ   1024
#define MROWS (BATCH*SEQ)     /* 16384 */
#define QKVCOLS (3*EMBED)     /* 2304  */

typedef __attribute__((ext_vector_type(16))) _Float16 v16h;
typedef __attribute__((ext_vector_type(8)))  _Float16 h8;
typedef __attribute__((ext_vector_type(4)))  _Float16 h4;
typedef __attribute__((ext_vector_type(8)))  float    v8f;

union Frag16 { v16h v; h8 h[2]; _Float16 e[16]; };

// ---------------------------------------------------------------------------
// CDNA5 async-tensor path: DMA 16B global -> LDS, tracked by ASYNCcnt.
// Generic pointers to __shared__ have the LDS byte offset in addr[31:0]
// (ISA 10.2 aperture mapping), so truncation gives the VDST LDS address.
// ---------------------------------------------------------------------------
__device__ __forceinline__ void async_b128(_Float16* lds_dst, const _Float16* gsrc) {
  uint32_t l = (uint32_t)(uintptr_t)lds_dst;
  asm volatile("global_load_async_to_lds_b128 %0, %1, off" :: "v"(l), "v"(gsrc) : "memory");
}
__device__ __forceinline__ void wait_async0() {
  asm volatile("s_wait_asynccnt 0x0" ::: "memory");
}

// A-fragment (16x32 f16): lane m=lane&15 holds row m; halves 0..7 = K[kb..kb+7],
// halves 8..15 = K[16+kb..16+kb+7], kb = (lane>=16)?8:0.  (ISA 7.12.2)
__device__ __forceinline__ v16h frag_a_lds(const _Float16* base, int lda, int lane) {
  const int row = lane & 15;
  const int kb  = (lane >> 4) * 8;
  Frag16 f;
  f.h[0] = *(const h8*)(base + row * lda + kb);
  f.h[1] = *(const h8*)(base + row * lda + 16 + kb);
  return f.v;
}

// B-fragment (32x16 f16) from an n-major tile T[n][k] (B[k][n] = T[n][k]).
__device__ __forceinline__ v16h frag_b_lds(const _Float16* base, int ldb, int lane) {
  const int col = lane & 15;
  const int kb  = (lane >> 4) * 8;
  Frag16 f;
  f.h[0] = *(const h8*)(base + col * ldb + kb);
  f.h[1] = *(const h8*)(base + col * ldb + 16 + kb);
  return f.v;
}

__device__ __forceinline__ v8f wmma_f16(v16h a, v16h b, v8f c) {
  return __builtin_amdgcn_wmma_f32_16x16x32_f16(false, a, false, b, (short)0, c, false, false);
}

// ---------------------------------------------------------------------------
// Kernel 0: f32 -> f16 conversion (x, w_qkv, w_proj), vectorized x4
// ---------------------------------------------------------------------------
__global__ __launch_bounds__(256) void mha_cvt_f16(const float* __restrict__ s,
                                                   _Float16* __restrict__ d, int n4) {
  const int i = blockIdx.x * 256 + threadIdx.x;
  if (i < n4) {
    float4 f = ((const float4*)s)[i];
    h4 o;
    o[0] = (_Float16)f.x; o[1] = (_Float16)f.y; o[2] = (_Float16)f.z; o[3] = (_Float16)f.w;
    ((h4*)d)[i] = o;
  }
}

// ---------------------------------------------------------------------------
// Shared GEMM main loop: C(128x128) = A(128xK) * B^T, A/B f16 row-major,
// async double-buffered LDS staging, 8 WMMA per wave per 32-K step.
// ---------------------------------------------------------------------------
__device__ __forceinline__ void gemm_mainloop(const _Float16* __restrict__ Ag,  // (m0,0)
                                              const _Float16* __restrict__ Bg,  // (n0,0)
                                              _Float16* As, _Float16* Bs,       // [2][128*40]
                                              int tid, int lane, int wrow, int wcol,
                                              v8f acc[2][4]) {
  auto stage = [&](int k0, int buf) {
    _Float16* a = As + buf * (128 * 40);
    _Float16* b = Bs + buf * (128 * 40);
#pragma unroll
    for (int t = 0; t < 2; ++t) {
      const int v   = tid + t * 256;   // 0..511
      const int row = v >> 2;          // 0..127
      const int c   = (v & 3) * 8;     // 0,8,16,24 halves
      async_b128(a + row * 40 + c, Ag + (size_t)row * EMBED + k0 + c);
      async_b128(b + row * 40 + c, Bg + (size_t)row * EMBED + k0 + c);
    }
  };

  stage(0, 0);
  wait_async0();
  __syncthreads();

  int c = 0;
  for (int k0 = 0; k0 < EMBED; k0 += 32, ++c) {
    const int cur = c & 1;
    if (k0 + 32 < EMBED) stage(k0 + 32, cur ^ 1);

    const _Float16* a = As + cur * (128 * 40);
    const _Float16* b = Bs + cur * (128 * 40);
    v16h a0 = frag_a_lds(a + (wrow * 32 + 0)  * 40, 40, lane);
    v16h a1 = frag_a_lds(a + (wrow * 32 + 16) * 40, 40, lane);
    v16h b0 = frag_b_lds(b + (wcol * 64 + 0)  * 40, 40, lane);
    v16h b1 = frag_b_lds(b + (wcol * 64 + 16) * 40, 40, lane);
    v16h b2 = frag_b_lds(b + (wcol * 64 + 32) * 40, 40, lane);
    v16h b3 = frag_b_lds(b + (wcol * 64 + 48) * 40, 40, lane);
    acc[0][0] = wmma_f16(a0, b0, acc[0][0]);
    acc[0][1] = wmma_f16(a0, b1, acc[0][1]);
    acc[0][2] = wmma_f16(a0, b2, acc[0][2]);
    acc[0][3] = wmma_f16(a0, b3, acc[0][3]);
    acc[1][0] = wmma_f16(a1, b0, acc[1][0]);
    acc[1][1] = wmma_f16(a1, b1, acc[1][1]);
    acc[1][2] = wmma_f16(a1, b2, acc[1][2]);
    acc[1][3] = wmma_f16(a1, b3, acc[1][3]);

    wait_async0();
    __syncthreads();
  }
}

// ---------------------------------------------------------------------------
// Kernel 1: qkv = x @ w_qkv^T (f16 in, f32 acc), scattered into Q/K/V buffers
// block: 256 threads (8 waves: 4 M x 2 N), tile 128x128, K-step 32
// ---------------------------------------------------------------------------
__global__ __launch_bounds__(256) void mha_qkv_gemm(const _Float16* __restrict__ Xh,
                                                    const _Float16* __restrict__ Wh,
                                                    _Float16* __restrict__ Qb,
                                                    _Float16* __restrict__ Kb,
                                                    _Float16* __restrict__ Vb) {
  __shared__ _Float16 As[2 * 128 * 40];
  __shared__ _Float16 Bs[2 * 128 * 40];
  const int tid  = threadIdx.x;
  const int lane = tid & 31;
  const int wave = tid >> 5;
  const int wrow = wave >> 1, wcol = wave & 1;
  const int n0 = blockIdx.x * 128;
  const int m0 = blockIdx.y * 128;

  v8f acc[2][4];
#pragma unroll
  for (int mt = 0; mt < 2; mt++)
#pragma unroll
    for (int nt = 0; nt < 4; nt++) acc[mt][nt] = (v8f){};

  gemm_mainloop(Xh + (size_t)m0 * EMBED, Wh + (size_t)n0 * EMBED,
                As, Bs, tid, lane, wrow, wcol, acc);

  // scatter: col e -> (which=e/768, head=(e%768)/64, d=e%64); row m -> (b,n)
  const int mbase = m0 + wrow * 32 + (lane >> 4) * 8;
  const int nbase = n0 + wcol * 64 + (lane & 15);
#pragma unroll
  for (int mt = 0; mt < 2; mt++)
#pragma unroll
    for (int nt = 0; nt < 4; nt++)
#pragma unroll
      for (int i = 0; i < 8; i++) {
        const int mg = mbase + mt * 16 + i;
        const int ng = nbase + nt * 16;
        const int which  = ng / EMBED;
        const int within = ng % EMBED;
        const int hh = within >> 6, dd = within & 63;
        const int bb = mg >> 10,    nn = mg & 1023;
        const size_t idx = (((size_t)bb * NHEAD + hh) * SEQ + nn) * HDIM + dd;
        const _Float16 val = (_Float16)acc[mt][nt][i];
        if (which == 0)      Qb[idx] = val;
        else if (which == 1) Kb[idx] = val;
        else                 Vb[idx] = val;
      }
}

// ---------------------------------------------------------------------------
// Kernel 2: flash attention per (b, h, 64-query tile); 128 threads = 4 waves.
// K chunks staged via async DMA, V chunks transposed, both double-buffered.
// ---------------------------------------------------------------------------
__global__ __launch_bounds__(128) void mha_flash_attn(const _Float16* __restrict__ Qb,
                                                      const _Float16* __restrict__ Kb,
                                                      const _Float16* __restrict__ Vb,
                                                      _Float16* __restrict__ Ob) {
  __shared__ _Float16 Ks[2 * 32 * 72];     // [key][d]  (n-major for S = Q*K^T)
  __shared__ _Float16 Vt[2 * 64 * 40];     // [d][key]  (n-major for O = P*V)
  __shared__ _Float16 Ps[4 * 16 * 40];     // per-wave P tile [16][32]

  const int tid  = threadIdx.x;
  const int lane = tid & 31;
  const int w    = tid >> 5;
  const int qt = blockIdx.x, hh = blockIdx.y, bb = blockIdx.z;

  const size_t hbase = (((size_t)bb * NHEAD + hh) * SEQ) * HDIM;
  const _Float16* Qh = Qb + hbase;
  const _Float16* Kh = Kb + hbase;
  const _Float16* Vh = Vb + hbase;
  const int q0 = qt * 64 + w * 16;

  // Q fragments in registers (dh=64 -> two 16x32 A-fragments)
  const _Float16* qp = Qh + (size_t)(q0 + (lane & 15)) * HDIM;
  const int kb = (lane >> 4) * 8;
  Frag16 qa0, qa1;
  qa0.h[0] = *(const h8*)(qp + kb);
  qa0.h[1] = *(const h8*)(qp + 16 + kb);
  qa1.h[0] = *(const h8*)(qp + 32 + kb);
  qa1.h[1] = *(const h8*)(qp + 48 + kb);

  float mi[8], li[8], alpha[8];
  v8f o[4];
#pragma unroll
  for (int dt = 0; dt < 4; dt++)
#pragma unroll
    for (int i = 0; i < 8; i++) o[dt][i] = 0.f;
#pragma unroll
  for (int i = 0; i < 8; i++) { mi[i] = -1e30f; li[i] = 0.f; }

  const float scale = 0.03608439182435161f;  // 1/sqrt(768)
  const int str = tid >> 2;         // 0..31  (key within chunk)
  const int stc = (tid & 3) * 16;   // 0,16,32,48 (d)

  _Float16* Pl = Ps + w * 16 * 40;
  const int prow = (lane >> 4) * 8;
  const int pcol = lane & 15;

  auto stage_k_async = [&](int kc, int buf) {
    _Float16* ks = Ks + buf * (32 * 72);
    const _Float16* gp = Kh + (size_t)(kc + str) * HDIM + stc;
    async_b128(ks + str * 72 + stc,     gp);
    async_b128(ks + str * 72 + stc + 8, gp + 8);
  };

  // prologue: chunk 0
  stage_k_async(0, 0);
  {
    const _Float16* gp = Vh + (size_t)str * HDIM + stc;
    h8 v0 = *(const h8*)gp;
    h8 v1 = *(const h8*)(gp + 8);
#pragma unroll
    for (int j = 0; j < 8; j++) Vt[(stc + j) * 40 + str]     = v0[j];
#pragma unroll
    for (int j = 0; j < 8; j++) Vt[(stc + 8 + j) * 40 + str] = v1[j];
  }
  wait_async0();
  __syncthreads();

  for (int c = 0; c < SEQ / 32; ++c) {
    const int cur = c & 1;
    const bool more = (c + 1) < (SEQ / 32);

    // issue next chunk: K via async DMA, V global loads (stores after compute)
    h8 nv0, nv1;
    if (more) {
      stage_k_async((c + 1) * 32, cur ^ 1);
      const _Float16* gp = Vh + (size_t)((c + 1) * 32 + str) * HDIM + stc;
      nv0 = *(const h8*)gp;
      nv1 = *(const h8*)(gp + 8);
    }

    const _Float16* ks = Ks + cur * (32 * 72);
    const _Float16* vt = Vt + cur * (64 * 40);

    // S chunk: 16 queries x 32 keys (two 16x16 tiles, dh=64 -> 2 k-steps each)
    v8f s0 = {}, s1 = {};
    s0 = wmma_f16(qa0.v, frag_b_lds(ks + 0 * 72 + 0,   72, lane), s0);
    s0 = wmma_f16(qa1.v, frag_b_lds(ks + 0 * 72 + 32,  72, lane), s0);
    s1 = wmma_f16(qa0.v, frag_b_lds(ks + 16 * 72 + 0,  72, lane), s1);
    s1 = wmma_f16(qa1.v, frag_b_lds(ks + 16 * 72 + 32, 72, lane), s1);

    // online softmax: row reductions across 16-lane half-wave groups
    float mnew[8];
#pragma unroll
    for (int i = 0; i < 8; i++) {
      const float a = s0[i] * scale;
      const float c2 = s1[i] * scale;
      s0[i] = a; s1[i] = c2;
      float v = fmaxf(a, c2);
      for (int off = 1; off < 16; off <<= 1) v = fmaxf(v, __shfl_xor(v, off, 32));
      mnew[i] = fmaxf(mi[i], v);
    }
#pragma unroll
    for (int i = 0; i < 8; i++) {
      const float p0 = __expf(s0[i] - mnew[i]);
      const float p1 = __expf(s1[i] - mnew[i]);
      float rs = p0 + p1;
      for (int off = 1; off < 16; off <<= 1) rs += __shfl_xor(rs, off, 32);
      alpha[i] = __expf(mi[i] - mnew[i]);
      li[i] = li[i] * alpha[i] + rs;
      mi[i] = mnew[i];
      // C-layout -> per-wave LDS (same-wave LDS ordering protects RAW)
      Pl[(prow + i) * 40 + pcol]      = (_Float16)p0;
      Pl[(prow + i) * 40 + 16 + pcol] = (_Float16)p1;
    }
#pragma unroll
    for (int dt = 0; dt < 4; dt++) {
      v8f t = o[dt];
#pragma unroll
      for (int i = 0; i < 8; i++) t[i] *= alpha[i];
      o[dt] = t;
    }

    // O += P (16x32) x V (32x64): 4 output d-tiles, single k-step
    v16h pa = frag_a_lds(Pl, 40, lane);
#pragma unroll
    for (int dt = 0; dt < 4; dt++)
      o[dt] = wmma_f16(pa, frag_b_lds(vt + dt * 16 * 40, 40, lane), o[dt]);

    // finish next-chunk V staging (transposed) into the other buffer
    if (more) {
      _Float16* vn = Vt + (cur ^ 1) * (64 * 40);
#pragma unroll
      for (int j = 0; j < 8; j++) vn[(stc + j) * 40 + str]     = nv0[j];
#pragma unroll
      for (int j = 0; j < 8; j++) vn[(stc + 8 + j) * 40 + str] = nv1[j];
    }

    wait_async0();
    __syncthreads();
  }

  // normalize and store f16 in [B, N, H*dh] layout for the projection GEMM
#pragma unroll
  for (int dt = 0; dt < 4; dt++)
#pragma unroll
    for (int i = 0; i < 8; i++) {
      const int q = q0 + prow + i;
      const size_t idx = ((size_t)(bb * SEQ + q)) * EMBED + hh * HDIM + dt * 16 + pcol;
      Ob[idx] = (_Float16)(o[dt][i] / li[i]);
    }
}

// ---------------------------------------------------------------------------
// Kernel 3: out = attn_out @ w_proj^T + b_proj (f16 in, f32 out)
// ---------------------------------------------------------------------------
__global__ __launch_bounds__(256) void mha_proj_gemm(const _Float16* __restrict__ A,
                                                     const _Float16* __restrict__ Wph,
                                                     const float* __restrict__ bias,
                                                     float* __restrict__ out) {
  __shared__ _Float16 As[2 * 128 * 40];
  __shared__ _Float16 Bs[2 * 128 * 40];
  const int tid  = threadIdx.x;
  const int lane = tid & 31;
  const int wave = tid >> 5;
  const int wrow = wave >> 1, wcol = wave & 1;
  const int n0 = blockIdx.x * 128;
  const int m0 = blockIdx.y * 128;

  v8f acc[2][4];
#pragma unroll
  for (int mt = 0; mt < 2; mt++)
#pragma unroll
    for (int nt = 0; nt < 4; nt++) acc[mt][nt] = (v8f){};

  gemm_mainloop(A + (size_t)m0 * EMBED, Wph + (size_t)n0 * EMBED,
                As, Bs, tid, lane, wrow, wcol, acc);

  const int mbase = m0 + wrow * 32 + (lane >> 4) * 8;
  const int nbase = n0 + wcol * 64 + (lane & 15);
#pragma unroll
  for (int mt = 0; mt < 2; mt++)
#pragma unroll
    for (int nt = 0; nt < 4; nt++)
#pragma unroll
      for (int i = 0; i < 8; i++) {
        const int mg = mbase + mt * 16 + i;
        const int ng = nbase + nt * 16;
        out[(size_t)mg * EMBED + ng] = acc[mt][nt][i] + bias[ng];
      }
}

extern "C" void kernel_launch(void* const* d_in, const int* in_sizes, int n_in,
                              void* d_out, int out_size, void* d_ws, size_t ws_size,
                              hipStream_t stream) {
  (void)in_sizes; (void)n_in; (void)out_size; (void)ws_size;
  const float* x      = (const float*)d_in[0];
  const float* w_qkv  = (const float*)d_in[1];
  const float* w_proj = (const float*)d_in[2];
  const float* b_proj = (const float*)d_in[3];
  float* out = (float*)d_out;

  const size_t elems = (size_t)BATCH * NHEAD * SEQ * HDIM;  // 12,582,912
  _Float16* Qb    = (_Float16*)d_ws;
  _Float16* Kb    = Qb + elems;
  _Float16* Vb    = Kb + elems;
  _Float16* Ob    = Vb + elems;                 // [B, N, 768] f16
  _Float16* Xh    = Ob + elems;                 // [16384, 768] f16
  _Float16* Wqkvh = Xh + (size_t)MROWS * EMBED; // [2304, 768] f16
  _Float16* Wph   = Wqkvh + (size_t)QKVCOLS * EMBED;  // [768, 768] f16
  // total ws use ~126 MB

  const int nx = MROWS * EMBED / 4, nw1 = QKVCOLS * EMBED / 4, nw2 = EMBED * EMBED / 4;
  mha_cvt_f16<<<dim3((nx  + 255) / 256), 256, 0, stream>>>(x,      Xh,    nx);
  mha_cvt_f16<<<dim3((nw1 + 255) / 256), 256, 0, stream>>>(w_qkv,  Wqkvh, nw1);
  mha_cvt_f16<<<dim3((nw2 + 255) / 256), 256, 0, stream>>>(w_proj, Wph,   nw2);

  mha_qkv_gemm<<<dim3(QKVCOLS / 128, MROWS / 128), 256, 0, stream>>>(Xh, Wqkvh, Qb, Kb, Vb);
  mha_flash_attn<<<dim3(SEQ / 64, NHEAD, BATCH), 128, 0, stream>>>(Qb, Kb, Vb, Ob);
  mha_proj_gemm<<<dim3(EMBED / 128, MROWS / 128), 256, 0, stream>>>(Ob, w_proj ? Wph : Wph, b_proj, out);
}
